// MultiHeadedAttention_28114855920343
// MI455X (gfx1250) — matrix operand order repaired and checked by reading, not compiled, not run on hardware
//
#include <hip/hip_runtime.h>
#include <stdint.h>

typedef __attribute__((ext_vector_type(16))) __bf16 v16bf;
typedef __attribute__((ext_vector_type(8)))  __bf16 v8bf;
typedef __attribute__((ext_vector_type(8)))  float  v8f;

constexpr int B_ = 4, S_ = 2048, D_ = 1024, H_ = 16, DK_ = 64;
// fold 1/sqrt(dk) and log2(e) into q so softmax runs in exp2 domain
constexpr float SCALE_LOG2E = 0.125f * 1.4426950408889634f;

__device__ inline v8f vzero8() {
  v8f v;
#pragma unroll
  for (int i = 0; i < 8; ++i) v[i] = 0.f;
  return v;
}

// ---- CDNA5 async global->LDS copy (ASYNCcnt-tracked), 16B per lane ----
__device__ inline void async_b128(uint32_t lds_byte_addr, uint64_t gaddr) {
  asm volatile("global_load_async_to_lds_b128 %0, %1, off"
               :: "v"(lds_byte_addr), "v"(gaddr)
               : "memory");
}
template <int N>
__device__ inline void wait_asynccnt() {
  asm volatile("s_wait_asynccnt %0" :: "i"(N) : "memory");
}
__device__ inline uint32_t lds_addr32(const void* p) {
  return (uint32_t)(uintptr_t)p;   // flat LDS addr: low 32 bits are the LDS offset
}

// A-operand fragment from GLOBAL (16x32 bf16): lane holds row; elements
// j<8 -> K = 8*hi + j ; j>=8 -> K = 16 + 8*hi + (j-8). Two contiguous 16B loads.
__device__ inline v16bf load_a16(const __bf16* rowk, int hi) {
  v8bf a = *(const v8bf*)(rowk + 8 * hi);
  v8bf b = *(const v8bf*)(rowk + 16 + 8 * hi);
  v16bf r;
#pragma unroll
  for (int i = 0; i < 8; ++i) { r[i] = a[i]; r[i + 8] = b[i]; }
  return r;
}
// A-operand fragment from LDS; rowbase points at this lane's row (+64*kk bytes)
__device__ inline v16bf lds_a16(const char* rowbase, int hi) {
  v8bf a = *(const v8bf*)(rowbase + 32 * hi);
  v8bf b = *(const v8bf*)(rowbase + 32 * hi + 16);
  v16bf r;
#pragma unroll
  for (int i = 0; i < 8; ++i) { r[i] = a[i]; r[i + 8] = b[i]; }
  return r;
}
// B-operand fragment from LDS: colbase -> 64B row of 32 K-values; two 16B loads (K = 8*hi+j, 16+8*hi+j)
__device__ inline v16bf lds_b16(const char* colbase, int hi) {
  v8bf a = *(const v8bf*)(colbase + 16 * hi);
  v8bf b = *(const v8bf*)(colbase + 32 + 16 * hi);
  v16bf r;
#pragma unroll
  for (int i = 0; i < 8; ++i) { r[i] = a[i]; r[i + 8] = b[i]; }
  return r;
}

__device__ inline v8f wmma_bf16(v16bf a, v16bf b, v8f c) {
  return __builtin_amdgcn_wmma_f32_16x16x32_bf16(false, a, false, b,
                                                 (short)0, c, false, false);
}

// ---------------- converters ----------------
__global__ void k_cvt(const float* __restrict__ in, __bf16* __restrict__ out, int n) {
  int i = blockIdx.x * blockDim.x + threadIdx.x;
  int stride = gridDim.x * blockDim.x;
  for (; i < n; i += stride) out[i] = (__bf16)in[i];
}
// out[z][c][r] = (bf16) in[z][r][c]
__global__ void k_cvt_t(const float* __restrict__ in, __bf16* __restrict__ out,
                        int R, int C) {
  size_t slab = (size_t)R * C;
  const float* pin = in + (size_t)blockIdx.z * slab;
  __bf16* pout = out + (size_t)blockIdx.z * slab;
  int i = blockIdx.x * blockDim.x + threadIdx.x;
  if (i < (int)slab) {
    int r = i / C, c = i % C;
    pout[(size_t)c * R + r] = (__bf16)pin[i];
  }
}

// ---------------- QKV projection ----------------
// grid (32, H, 3). Block: 8 waves, each owns 2 row-tiles (32 rows), all share the
// 64-column weight panel of head blockIdx.y. The 32-deep K-slab of the panel
// (4KB) is async-staged into LDS, double buffered.
__global__ void __launch_bounds__(256) k_qkv(
    const __bf16* __restrict__ xb,
    const __bf16* __restrict__ wqt, const __bf16* __restrict__ wkt,
    const __bf16* __restrict__ wvt,
    __bf16* __restrict__ q, __bf16* __restrict__ kmat, __bf16* __restrict__ vt) {
  __shared__ __align__(32) char sB[2][4096];   // [col 0..63][32 K-vals * 2B]
  const int lane = threadIdx.x & 31, lo = lane & 15, hi = lane >> 4;
  const int w = threadIdx.x >> 5, t = threadIdx.x;
  const int mgroup = blockIdx.x, h = blockIdx.y, which = blockIdx.z;
  const __bf16* wt = (which == 0) ? wqt : (which == 1) ? wkt : wvt;

  // staging: col = t/4 (row of wt, 2KB apart), 16B chunk (t&3)
  const uint64_t wsrc = (uint64_t)(uintptr_t)(wt + (size_t)h * DK_ * D_) +
                        (uint64_t)(t >> 2) * (D_ * 2) + (uint64_t)(t & 3) * 16;
  const uint32_t sdst[2] = { lds_addr32(&sB[0][0]) + (uint32_t)t * 16,
                             lds_addr32(&sB[1][0]) + (uint32_t)t * 16 };

  const int m0 = (mgroup * 16 + w * 2) * 16;       // first row of this wave
  const __bf16* arow0 = xb + (size_t)(m0 + lo) * D_;
  const __bf16* arow1 = xb + (size_t)(m0 + 16 + lo) * D_;

  v8f acc[2][4];
#pragma unroll
  for (int mi = 0; mi < 2; ++mi)
#pragma unroll
    for (int tt = 0; tt < 4; ++tt) acc[mi][tt] = vzero8();

  async_b128(sdst[0], wsrc);                        // prologue: slab 0 -> buf 0
  for (int kb = 0, ib = 0; kb < D_; kb += 32, ++ib) {
    const int cur = ib & 1;
    if (kb + 32 < D_) {
      async_b128(sdst[cur ^ 1], wsrc + (uint64_t)(kb + 32) * 2);
      wait_asynccnt<1>();
    } else {
      wait_asynccnt<0>();
    }
    __syncthreads();                                // slab visible to all waves
    const char* bufB = sB[cur];
    v16bf a0 = load_a16(arow0 + kb, hi);
    v16bf a1 = load_a16(arow1 + kb, hi);
#pragma unroll
    for (int tt = 0; tt < 4; ++tt) {
      v16bf b = lds_b16(bufB + (16 * tt + lo) * 64, hi);
      acc[0][tt] = wmma_bf16(a0, b, acc[0][tt]);
      acc[1][tt] = wmma_bf16(a1, b, acc[1][tt]);
    }
    __syncthreads();                                // reads done before overwrite
  }

#pragma unroll
  for (int mi = 0; mi < 2; ++mi)
#pragma unroll
    for (int tt = 0; tt < 4; ++tt)
#pragma unroll
      for (int v = 0; v < 8; ++v) {
        int gr = m0 + mi * 16 + v + 8 * hi;
        int bb = gr / S_, s = gr % S_;
        int n = 16 * tt + lo;
        float val = acc[mi][tt][v];
        size_t bh = (size_t)bb * H_ + h;
        if (which == 0)
          q[(bh * S_ + s) * DK_ + n] = (__bf16)(val * SCALE_LOG2E);
        else if (which == 1)
          kmat[(bh * S_ + s) * DK_ + n] = (__bf16)val;
        else
          vt[(bh * DK_ + n) * S_ + s] = (__bf16)val;  // transposed [B,H,Dv,S]
      }
}

// ---------------- flash attention ----------------
// grid (8, B*H). Block: 8 waves x 2 query tiles = 256 queries of one (b,h).
// 32-key K/V tiles (4KB each) async-staged into LDS, double buffered, shared
// by all 8 waves. Scores computed transposed (S^T = K*Q^T) so the f32 C-tile
// layout matches the bf16 A-fragment layout of P element-for-element.
__global__ void __launch_bounds__(256) k_flash(
    const __bf16* __restrict__ qg, const __bf16* __restrict__ kg,
    const __bf16* __restrict__ vg, __bf16* __restrict__ attn) {
  __shared__ __align__(32) char sK[2][4096];   // [key 0..31][64 dims * 2B]
  __shared__ __align__(32) char sV[2][4096];   // [dv  0..63][32 keys * 2B]
  const int lane = threadIdx.x & 31, lo = lane & 15, hi = lane >> 4;
  const int w = threadIdx.x >> 5, t = threadIdx.x;
  const int qgroup = blockIdx.x, bh = blockIdx.y;
  const int qt0 = qgroup * 16 + w * 2;

  const __bf16* qb = qg + ((size_t)bh * S_ + qt0 * 16) * DK_;
  const uint64_t kbase = (uint64_t)(uintptr_t)(kg + (size_t)bh * S_ * DK_);
  const uint64_t vbase = (uint64_t)(uintptr_t)(vg + (size_t)bh * DK_ * S_);
  const uint64_t ksrc = kbase + (uint64_t)(t >> 3) * (DK_ * 2) + (uint64_t)(t & 7) * 16;
  const uint64_t vsrc = vbase + (uint64_t)(t >> 2) * (S_ * 2) + (uint64_t)(t & 3) * 16;
  const uint32_t kdst[2] = { lds_addr32(&sK[0][0]) + (uint32_t)t * 16,
                             lds_addr32(&sK[1][0]) + (uint32_t)t * 16 };
  const uint32_t vdst[2] = { lds_addr32(&sV[0][0]) + (uint32_t)t * 16,
                             lds_addr32(&sV[1][0]) + (uint32_t)t * 16 };

  // Q as B-operand (N = queries), two K-halves of dk=64, loaded once
  v16bf qf[2][2];
#pragma unroll
  for (int qi = 0; qi < 2; ++qi)
#pragma unroll
    for (int kk = 0; kk < 2; ++kk)
      qf[qi][kk] = *(const v16bf*)(qb + (size_t)(qi * 16 + lo) * DK_ + 32 * kk + 16 * hi);

  v8f O[2][4];
#pragma unroll
  for (int qi = 0; qi < 2; ++qi)
#pragma unroll
    for (int tt = 0; tt < 4; ++tt) O[qi][tt] = vzero8();
  float m[2] = { -3.0e38f, -3.0e38f }, l[2] = { 0.f, 0.f };

  async_b128(kdst[0], ksrc);                       // prologue: block 0 -> buf 0
  async_b128(vdst[0], vsrc);
  for (int kb = 0, ib = 0; kb < S_; kb += 32, ++ib) {
    const int cur = ib & 1;
    if (kb + 32 < S_) {
      async_b128(kdst[cur ^ 1], ksrc + (uint64_t)(kb + 32) * (DK_ * 2));
      async_b128(vdst[cur ^ 1], vsrc + (uint64_t)(kb + 32) * 2);
      wait_asynccnt<2>();
    } else {
      wait_asynccnt<0>();
    }
    __syncthreads();
    const char* bK = sK[cur];
    const char* bV = sV[cur];

    // S^T tiles: T0 = keys kb..kb+15, T1 = keys kb+16..kb+31 (M=key, N=query)
    v8f s0[2] = { vzero8(), vzero8() }, s1[2] = { vzero8(), vzero8() };
#pragma unroll
    for (int kk = 0; kk < 2; ++kk) {
      v16bf ka0 = lds_a16(bK + (lo) * 128 + kk * 64, hi);
      v16bf ka1 = lds_a16(bK + (16 + lo) * 128 + kk * 64, hi);
#pragma unroll
      for (int qi = 0; qi < 2; ++qi) {
        s0[qi] = wmma_bf16(ka0, qf[qi][kk], s0[qi]);
        s1[qi] = wmma_bf16(ka1, qf[qi][kk], s1[qi]);
      }
    }

    v16bf pf[2];
#pragma unroll
    for (int qi = 0; qi < 2; ++qi) {
      float mx = -3.0e38f;
#pragma unroll
      for (int v = 0; v < 8; ++v) { mx = fmaxf(mx, s0[qi][v]); mx = fmaxf(mx, s1[qi][v]); }
      mx = fmaxf(mx, __shfl_xor(mx, 16, 32));
      float mn = fmaxf(m[qi], mx);
      float corr = __builtin_amdgcn_exp2f(m[qi] - mn);
      float p0[8], p1[8], rs = 0.f;
#pragma unroll
      for (int v = 0; v < 8; ++v) {
        p0[v] = __builtin_amdgcn_exp2f(s0[qi][v] - mn);
        p1[v] = __builtin_amdgcn_exp2f(s1[qi][v] - mn);
        rs += p0[v] + p1[v];
      }
      rs += __shfl_xor(rs, 16, 32);
      l[qi] = l[qi] * corr + rs;
      m[qi] = mn;
#pragma unroll
      for (int v = 0; v < 8; ++v) { pf[qi][v] = (__bf16)p0[v]; pf[qi][v + 8] = (__bf16)p1[v]; }
#pragma unroll
      for (int v = 0; v < 8; ++v) {
        float cv = __shfl(corr, v + 8 * hi, 32);
        O[qi][0][v] *= cv; O[qi][1][v] *= cv; O[qi][2][v] *= cv; O[qi][3][v] *= cv;
      }
    }

#pragma unroll
    for (int tt = 0; tt < 4; ++tt) {
      v16bf vf = lds_b16(bV + (16 * tt + lo) * 64, hi);
      O[0][tt] = wmma_bf16(pf[0], vf, O[0][tt]);
      O[1][tt] = wmma_bf16(pf[1], vf, O[1][tt]);
    }
    __syncthreads();
  }

  const int bb = bh / H_, h = bh % H_;
#pragma unroll
  for (int qi = 0; qi < 2; ++qi)
#pragma unroll
    for (int v = 0; v < 8; ++v) {
      float lv = __shfl(l[qi], v + 8 * hi, 32);
      float inv = 1.0f / lv;
      int s = (qt0 + qi) * 16 + v + 8 * hi;
#pragma unroll
      for (int tt = 0; tt < 4; ++tt) {
        int dv = 16 * tt + lo;
        attn[((size_t)bb * S_ + s) * (H_ * DK_) + h * DK_ + dv] = (__bf16)(O[qi][tt][v] * inv);
      }
    }
}

// ---------------- output projection ----------------
// grid (32, 16). Block: 8 waves x 2 row-tiles share one 64-column W_O^T panel,
// K-slabs async-staged in LDS (same pattern as k_qkv). fp32 output.
__global__ void __launch_bounds__(256) k_oproj(
    const __bf16* __restrict__ attn, const __bf16* __restrict__ wot,
    float* __restrict__ out) {
  __shared__ __align__(32) char sB[2][4096];
  const int lane = threadIdx.x & 31, lo = lane & 15, hi = lane >> 4;
  const int w = threadIdx.x >> 5, t = threadIdx.x;
  const int mgroup = blockIdx.x, n64 = blockIdx.y;

  const uint64_t wsrc = (uint64_t)(uintptr_t)(wot + (size_t)n64 * 64 * D_) +
                        (uint64_t)(t >> 2) * (D_ * 2) + (uint64_t)(t & 3) * 16;
  const uint32_t sdst[2] = { lds_addr32(&sB[0][0]) + (uint32_t)t * 16,
                             lds_addr32(&sB[1][0]) + (uint32_t)t * 16 };

  const int m0 = (mgroup * 16 + w * 2) * 16;
  const __bf16* arow0 = attn + (size_t)(m0 + lo) * D_;
  const __bf16* arow1 = attn + (size_t)(m0 + 16 + lo) * D_;

  v8f acc[2][4];
#pragma unroll
  for (int mi = 0; mi < 2; ++mi)
#pragma unroll
    for (int tt = 0; tt < 4; ++tt) acc[mi][tt] = vzero8();

  async_b128(sdst[0], wsrc);
  for (int kb = 0, ib = 0; kb < D_; kb += 32, ++ib) {
    const int cur = ib & 1;
    if (kb + 32 < D_) {
      async_b128(sdst[cur ^ 1], wsrc + (uint64_t)(kb + 32) * 2);
      wait_asynccnt<1>();
    } else {
      wait_asynccnt<0>();
    }
    __syncthreads();
    const char* bufB = sB[cur];
    v16bf a0 = load_a16(arow0 + kb, hi);
    v16bf a1 = load_a16(arow1 + kb, hi);
#pragma unroll
    for (int tt = 0; tt < 4; ++tt) {
      v16bf b = lds_b16(bufB + (16 * tt + lo) * 64, hi);
      acc[0][tt] = wmma_bf16(a0, b, acc[0][tt]);
      acc[1][tt] = wmma_bf16(a1, b, acc[1][tt]);
    }
    __syncthreads();
  }
#pragma unroll
  for (int mi = 0; mi < 2; ++mi)
#pragma unroll
    for (int tt = 0; tt < 4; ++tt)
#pragma unroll
      for (int v = 0; v < 8; ++v)
        out[(size_t)(m0 + mi * 16 + v + 8 * hi) * D_ + n64 * 64 + 16 * tt + lo] =
            acc[mi][tt][v];
}

extern "C" void kernel_launch(void* const* d_in, const int* in_sizes, int n_in,
                              void* d_out, int out_size, void* d_ws, size_t ws_size,
                              hipStream_t stream) {
  (void)in_sizes; (void)n_in; (void)out_size; (void)ws_size;
  const float* x  = (const float*)d_in[0];
  const float* WQ = (const float*)d_in[1];
  const float* WK = (const float*)d_in[2];
  const float* WV = (const float*)d_in[3];
  const float* WO = (const float*)d_in[4];

  char* ws = (char*)d_ws;
  size_t off = 0;
  auto take = [&](size_t nelem) { char* p = ws + off; off += nelem * sizeof(__bf16); return (__bf16*)p; };
  __bf16* xb   = take((size_t)B_ * S_ * D_);
  __bf16* q    = take((size_t)B_ * H_ * S_ * DK_);
  __bf16* kmat = take((size_t)B_ * H_ * S_ * DK_);
  __bf16* vt   = take((size_t)B_ * H_ * DK_ * S_);
  __bf16* attn = take((size_t)B_ * S_ * D_);
  __bf16* wqt  = take((size_t)H_ * D_ * DK_);
  __bf16* wkt  = take((size_t)H_ * D_ * DK_);
  __bf16* wvt  = take((size_t)H_ * D_ * DK_);
  __bf16* wot  = take((size_t)D_ * D_);

  k_cvt<<<4096, 256, 0, stream>>>(x, xb, B_ * S_ * D_);
  k_cvt_t<<<dim3((D_ * DK_ + 255) / 256, 1, H_), 256, 0, stream>>>(WQ, wqt, D_, DK_);
  k_cvt_t<<<dim3((D_ * DK_ + 255) / 256, 1, H_), 256, 0, stream>>>(WK, wkt, D_, DK_);
  k_cvt_t<<<dim3((D_ * DK_ + 255) / 256, 1, H_), 256, 0, stream>>>(WV, wvt, D_, DK_);
  k_cvt_t<<<dim3((D_ * D_ + 255) / 256, 1, 1), 256, 0, stream>>>(WO, wot, D_, D_);

  k_qkv<<<dim3(32, H_, 3), 256, 0, stream>>>(xb, wqt, wkt, wvt, q, kmat, vt);
  k_flash<<<dim3(8, B_ * H_), 256, 0, stream>>>(q, kmat, vt, attn);
  k_oproj<<<dim3(32, 16), 256, 0, stream>>>(attn, wot, (float*)d_out);
}
